// RGIN_8280696947364
// MI455X (gfx1250) — matrix operand order repaired and checked by reading, not compiled, not run on hardware
//
#include <hip/hip_runtime.h>

// Problem constants (from reference)
#define NN 50000
#define NE 800000
#define DD 64
#define NL 3
#define NR 2
#define NG 128
#define NC 10
#define TILES 3125   // NN / 16 exactly

typedef _Float16 v16h __attribute__((ext_vector_type(16)));
typedef _Float16 h8   __attribute__((ext_vector_type(8)));
typedef float    v8f  __attribute__((ext_vector_type(8)));

static __device__ __forceinline__ v8f wmma16(v16h a, v16h b, v8f c) {
  // D = A(16x32 f16) * B(32x16 f16) + C(16x16 f32)
  return __builtin_amdgcn_wmma_f32_16x16x32_f16(false, a, false, b, (short)0, c,
                                                false, false);
}

static __device__ __forceinline__ void atomic_add_f32(float* p, float v) {
  __hip_atomic_fetch_add(p, v, __ATOMIC_RELAXED, __HIP_MEMORY_SCOPE_AGENT);
}

// ---------------------------------------------------------------- zero fill
__global__ void zero_kernel(float* __restrict__ p, long n) {
  long i = (long)blockIdx.x * blockDim.x + threadIdx.x;
  if (i < n) p[i] = 0.0f;
}

// ----------------------------------------------- edge scatter (one pass / layer)
// thread = (edge, 4-float chunk); agg[r][dst][d] += h[src][d]
__global__ __launch_bounds__(256) void scatter_kernel(
    const float* __restrict__ h, const int* __restrict__ ei,
    const int* __restrict__ et, float* __restrict__ agg) {
  long tid = (long)blockIdx.x * 256 + threadIdx.x;
  if (tid >= (long)NE * 16) return;
  int e = (int)(tid >> 4);
  int q = (int)(tid & 15);
  int s = ei[e];
  int d = ei[NE + e];
  int r = et[e];
  float4 hv = *(const float4*)(h + (size_t)s * DD + q * 4);
  float* ap = agg + ((size_t)r * NN + d) * DD + q * 4;
  atomic_add_f32(ap + 0, hv.x);
  atomic_add_f32(ap + 1, hv.y);
  atomic_add_f32(ap + 2, hv.z);
  atomic_add_f32(ap + 3, hv.w);
}

// ------------------------------------------------------- per-layer node kernel
// One wave owns a 16-row node tile; computes
//   h_out = h@Wself + bself + sum_r relu(((1+eps_r)h + agg_r)@W1_r + b1_r)@W2_r + b2_r
// All 5 weight matrices live transposed (n-major) as f16 in LDS.
__global__ __launch_bounds__(256) void node_kernel(
    const float* __restrict__ h_in, const float* __restrict__ agg,
    const float* __restrict__ selfW, const float* __restrict__ selfB,
    const float* __restrict__ epsv, const float* __restrict__ W1,
    const float* __restrict__ B1, const float* __restrict__ W2,
    const float* __restrict__ B2, float* __restrict__ h_out) {
  __shared__ __align__(32) _Float16 wT[5][DD * DD];   // [n][k] transposed, f16
  __shared__ __align__(32) _Float16 zscr[8][16 * DD]; // per-wave D->A transpose

  {  // cooperative f32 -> f16 transposed weight staging
    const float* srcs[5] = {selfW, W1, W2, W1 + DD * DD, W2 + DD * DD};
    for (int idx = threadIdx.x; idx < 5 * DD * DD; idx += 256) {
      int mi = idx >> 12;
      int e = idx & (DD * DD - 1);
      int k = e >> 6, n = e & 63;
      wT[mi][n * DD + k] = (_Float16)srcs[mi][e];
    }
  }
  __syncthreads();

  const int wid = threadIdx.x >> 5;
  const int lane = threadIdx.x & 31;
  const int mcol = lane & 15;  // A row / C,D column within tile
  const int hl = lane >> 4;    // half-wave select
  const int tile = blockIdx.x * 8 + wid;
  if (tile >= TILES) return;
  const int node = tile * 16 + mcol;

  // h tile as two 16x32 A-fragments; keep f32 copy for z = (1+eps)h + agg
  float h32[2][16];
  v16h aF[2];
#pragma unroll
  for (int kb = 0; kb < 2; ++kb) {
#pragma unroll
    for (int t = 0; t < 2; ++t) {
      const float4* p =
          (const float4*)(h_in + (size_t)node * DD + kb * 32 + t * 8 + hl * 16);
      float4 a = p[0], b = p[1];
      int e0 = t * 8;
      h32[kb][e0 + 0] = a.x; h32[kb][e0 + 1] = a.y;
      h32[kb][e0 + 2] = a.z; h32[kb][e0 + 3] = a.w;
      h32[kb][e0 + 4] = b.x; h32[kb][e0 + 5] = b.y;
      h32[kb][e0 + 6] = b.z; h32[kb][e0 + 7] = b.w;
    }
#pragma unroll
    for (int e = 0; e < 16; ++e) aF[kb][e] = (_Float16)h32[kb][e];
  }

  // x_new = h @ Wself + bself
  v8f acc[4];
#pragma unroll
  for (int nb = 0; nb < 4; ++nb) {
    float bv = selfB[nb * 16 + mcol];
#pragma unroll
    for (int i = 0; i < 8; ++i) acc[nb][i] = bv;
#pragma unroll
    for (int kb = 0; kb < 2; ++kb) {
      v16h bF = *(const v16h*)(&wT[0][(nb * 16 + mcol) * DD + kb * 32 + hl * 16]);
      acc[nb] = wmma16(aF[kb], bF, acc[nb]);
    }
  }

#pragma unroll
  for (int r = 0; r < NR; ++r) {
    const float ep = 1.0f + epsv[r];
    const int w1s = 1 + 2 * r;
    const int w2s = 2 + 2 * r;

    // z-input fragments: (1+eps)*h + agg_r, converted to f16 A layout
    v16h zF[2];
#pragma unroll
    for (int kb = 0; kb < 2; ++kb) {
#pragma unroll
      for (int t = 0; t < 2; ++t) {
        const float4* p = (const float4*)(agg + ((size_t)r * NN + node) * DD +
                                          kb * 32 + t * 8 + hl * 16);
        float4 a = p[0], b = p[1];
        int e0 = t * 8;
        zF[kb][e0 + 0] = (_Float16)(ep * h32[kb][e0 + 0] + a.x);
        zF[kb][e0 + 1] = (_Float16)(ep * h32[kb][e0 + 1] + a.y);
        zF[kb][e0 + 2] = (_Float16)(ep * h32[kb][e0 + 2] + a.z);
        zF[kb][e0 + 3] = (_Float16)(ep * h32[kb][e0 + 3] + a.w);
        zF[kb][e0 + 4] = (_Float16)(ep * h32[kb][e0 + 4] + b.x);
        zF[kb][e0 + 5] = (_Float16)(ep * h32[kb][e0 + 5] + b.y);
        zF[kb][e0 + 6] = (_Float16)(ep * h32[kb][e0 + 6] + b.z);
        zF[kb][e0 + 7] = (_Float16)(ep * h32[kb][e0 + 7] + b.w);
      }
    }

    // zmid = z @ W1 + b1
    v8f zacc[4];
#pragma unroll
    for (int nb = 0; nb < 4; ++nb) {
      float bv = B1[r * DD + nb * 16 + mcol];
#pragma unroll
      for (int i = 0; i < 8; ++i) zacc[nb][i] = bv;
#pragma unroll
      for (int kb = 0; kb < 2; ++kb) {
        v16h bF =
            *(const v16h*)(&wT[w1s][(nb * 16 + mcol) * DD + kb * 32 + hl * 16]);
        zacc[nb] = wmma16(zF[kb], bF, zacc[nb]);
      }
    }

    // relu + D-layout -> A-layout transition through per-wave LDS scratch
#pragma unroll
    for (int nb = 0; nb < 4; ++nb)
#pragma unroll
      for (int v = 0; v < 8; ++v)
        zscr[wid][(v + 8 * hl) * DD + nb * 16 + mcol] =
            (_Float16)fmaxf(zacc[nb][v], 0.0f);
    __threadfence_block();  // DS in-order within wave; force dscnt wait

    v16h z2F[2];
#pragma unroll
    for (int kb = 0; kb < 2; ++kb) {
#pragma unroll
      for (int t = 0; t < 2; ++t) {
        h8 rr = *(const h8*)(&zscr[wid][mcol * DD + kb * 32 + t * 8 + hl * 16]);
        int e0 = t * 8;
#pragma unroll
        for (int e = 0; e < 8; ++e) z2F[kb][e0 + e] = rr[e];
      }
    }

    // x_new += zmid @ W2 + b2  (accumulate straight into acc as WMMA C)
#pragma unroll
    for (int nb = 0; nb < 4; ++nb) {
      float bv = B2[r * DD + nb * 16 + mcol];
#pragma unroll
      for (int i = 0; i < 8; ++i) acc[nb][i] += bv;
#pragma unroll
      for (int kb = 0; kb < 2; ++kb) {
        v16h bF =
            *(const v16h*)(&wT[w2s][(nb * 16 + mcol) * DD + kb * 32 + hl * 16]);
        acc[nb] = wmma16(z2F[kb], bF, acc[nb]);
      }
    }
    __threadfence_block();  // protect zscr reuse across r iterations
  }

  // store the 16x64 tile (D layout: lane -> column, VGPR -> row)
#pragma unroll
  for (int nb = 0; nb < 4; ++nb)
#pragma unroll
    for (int v = 0; v < 8; ++v)
      h_out[((size_t)tile * 16 + v + 8 * hl) * DD + nb * 16 + mcol] = acc[nb][v];
}

// ------------------------------------------------------------- pooling
__global__ __launch_bounds__(256) void pool_sum_kernel(
    const float* __restrict__ h, const int* __restrict__ batch,
    float* __restrict__ sums) {
  long tid = (long)blockIdx.x * 256 + threadIdx.x;
  if (tid >= (long)NN * 16) return;
  int n = (int)(tid >> 4);
  int q = (int)(tid & 15);
  int g = batch[n];
  float4 v = *(const float4*)(h + (size_t)n * DD + q * 4);
  float* sp = sums + (size_t)g * DD + q * 4;
  atomic_add_f32(sp + 0, v.x);
  atomic_add_f32(sp + 1, v.y);
  atomic_add_f32(sp + 2, v.z);
  atomic_add_f32(sp + 3, v.w);
}

__global__ void pool_cnt_kernel(const int* __restrict__ batch,
                                float* __restrict__ counts) {
  int n = blockIdx.x * blockDim.x + threadIdx.x;
  if (n < NN) atomic_add_f32(counts + batch[n], 1.0f);
}

// ------------------------------------------------------------- MLP head
__global__ void head_kernel(const float* __restrict__ sums,
                            const float* __restrict__ counts,
                            const float* __restrict__ lin1W,
                            const float* __restrict__ lin1B,
                            const float* __restrict__ lin2W,
                            const float* __restrict__ lin2B,
                            float* __restrict__ out) {
  int g = threadIdx.x;
  if (g >= NG) return;
  float inv = 1.0f / fmaxf(counts[g], 1.0f);
  float pooled[DD];
#pragma unroll
  for (int k = 0; k < DD; ++k) pooled[k] = sums[g * DD + k] * inv;
  float oc[NC];
#pragma unroll
  for (int c = 0; c < NC; ++c) oc[c] = lin2B[c];
  for (int j = 0; j < DD; ++j) {
    float t = lin1B[j];
#pragma unroll
    for (int k = 0; k < DD; ++k) t += pooled[k] * lin1W[k * DD + j];
    t = fmaxf(t, 0.0f);
#pragma unroll
    for (int c = 0; c < NC; ++c) oc[c] += t * lin2W[j * NC + c];
  }
#pragma unroll
  for (int c = 0; c < NC; ++c) out[g * NC + c] = oc[c];
}

// ------------------------------------------------------------- launcher
extern "C" void kernel_launch(void* const* d_in, const int* in_sizes, int n_in,
                              void* d_out, int out_size, void* d_ws,
                              size_t ws_size, hipStream_t stream) {
  (void)in_sizes; (void)n_in; (void)out_size; (void)ws_size;

  const float* x     = (const float*)d_in[0];
  const int*   ei    = (const int*)d_in[1];
  const int*   et    = (const int*)d_in[2];
  const int*   batch = (const int*)d_in[3];
  const float* selfW = (const float*)d_in[4];
  const float* selfB = (const float*)d_in[5];
  const float* eps   = (const float*)d_in[6];
  const float* W1    = (const float*)d_in[7];
  const float* B1    = (const float*)d_in[8];
  const float* W2    = (const float*)d_in[9];
  const float* B2    = (const float*)d_in[10];
  const float* l1W   = (const float*)d_in[11];
  const float* l1B   = (const float*)d_in[12];
  const float* l2W   = (const float*)d_in[13];
  const float* l2B   = (const float*)d_in[14];

  // workspace layout (f32): hA | hB | agg[R][N][D] | sums[G][D] | counts[G]
  float* hA     = (float*)d_ws;
  float* hB     = hA + (size_t)NN * DD;
  float* agg    = hB + (size_t)NN * DD;
  float* sums   = agg + (size_t)NR * NN * DD;
  float* counts = sums + (size_t)NG * DD;   // contiguous after sums

  const long aggN = (long)NR * NN * DD;
  const long poolN = (long)NG * DD + NG;

  const float* hin = x;
  float* houts[NL] = {hA, hB, hA};
  for (int l = 0; l < NL; ++l) {
    zero_kernel<<<(int)((aggN + 255) / 256), 256, 0, stream>>>(agg, aggN);
    scatter_kernel<<<(int)(((long)NE * 16 + 255) / 256), 256, 0, stream>>>(
        hin, ei, et, agg);
    node_kernel<<<(TILES + 7) / 8, 256, 0, stream>>>(
        hin, agg, selfW + (size_t)l * DD * DD, selfB + (size_t)l * DD,
        eps + (size_t)l * NR, W1 + (size_t)l * NR * DD * DD,
        B1 + (size_t)l * NR * DD, W2 + (size_t)l * NR * DD * DD,
        B2 + (size_t)l * NR * DD, houts[l]);
    hin = houts[l];
  }

  zero_kernel<<<(int)((poolN + 255) / 256), 256, 0, stream>>>(sums, poolN);
  pool_sum_kernel<<<(int)(((long)NN * 16 + 255) / 256), 256, 0, stream>>>(
      hin, batch, sums);
  pool_cnt_kernel<<<(NN + 255) / 256, 256, 0, stream>>>(batch, counts);
  head_kernel<<<1, NG, 0, stream>>>(sums, counts, l1W, l1B, l2W, l2B,
                                    (float*)d_out);
}
